// SAConvLSTM_51780125720614
// MI455X (gfx1250) — matrix operand order repaired and checked by reading, not compiled, not run on hardware
//
#include <hip/hip_runtime.h>
#include <hip/hip_bf16.h>
#include <math.h>

// dims
#define BB    16
#define TT    16
#define HWN   1024
#define CCAT  128          // CIN + COUT
#define G4    256          // 4*C_OUT
#define KCONV 1152         // CCAT*9
#define NCHUNK 36          // KCONV/32

typedef __bf16 bf16;
typedef __attribute__((ext_vector_type(16))) __bf16 bf16x16;
typedef __attribute__((ext_vector_type(8)))  float  f32x8;

// A-fragment K order: stored slot s holds logical K kperm(s) = s (contiguous,
// matching the contiguous B fragments).
static __device__ __forceinline__ int kperm(int s) {
  return s;
}

static __device__ __forceinline__ f32x8 wmma_bf16(bf16x16 a, bf16x16 b, f32x8 c) {
  return __builtin_amdgcn_wmma_f32_16x16x32_bf16(false, a, false, b, (short)0, c, false, false);
}
static __device__ __forceinline__ bf16x16 ldfrag(const bf16* p) {
  return *(const bf16x16*)p;   // 32B, all call sites 32B-aligned by construction
}
static __device__ __forceinline__ float sigm(float x) { return 1.f / (1.f + __expf(-x)); }

// ---------------- one-time prepacks ----------------

// conv_w (256,128,3,3) f32 -> wpack[((ct*36+chunk)*16+m)*32+s] bf16, A-fragment order.
// K ordering: k = tap*128 + cin  (tap = dy*3+dx), so each 32-chunk is one tap.
__global__ __launch_bounds__(256) void k_prepack_convw(const float* __restrict__ conv_w,
                                                       bf16* __restrict__ wpack) {
  int i = blockIdx.x * 256 + threadIdx.x;     // < 36*256*32
  int s = i & 31;
  int r = i >> 5;
  int m = r & 15;
  int r2 = r >> 4;                            // ct*36 + chunk
  int chunk = r2 % NCHUNK;
  int ct = r2 / NCHUNK;
  int cout = ct * 16 + m;
  int k = chunk * 32 + kperm(s);
  int tap = k >> 7;
  int cin = k & 127;
  wpack[i] = (bf16)conv_w[(size_t)cout * KCONV + (size_t)cin * 9 + tap];
}

// q_w/k_w (16,64), v_w/z_w (64,64) -> A-fragment packs: [((ot*2+chunk)*16+m)*32+s]
__global__ __launch_bounds__(256) void k_prepack_small(const float* __restrict__ qw,
                                                       const float* __restrict__ kw,
                                                       const float* __restrict__ vw,
                                                       const float* __restrict__ zw,
                                                       bf16* __restrict__ apq, bf16* __restrict__ apk,
                                                       bf16* __restrict__ apv, bf16* __restrict__ apz) {
  int i = blockIdx.x * 256 + threadIdx.x;     // < 10240
  if (i >= 10240) return;
  const float* w; bf16* dst; int j;
  if (i < 1024)      { w = qw; dst = apq; j = i; }
  else if (i < 2048) { w = kw; dst = apk; j = i - 1024; }
  else if (i < 6144) { w = vw; dst = apv; j = i - 2048; }
  else               { w = zw; dst = apz; j = i - 6144; }
  int s = j & 31, r = j >> 5, m = r & 15, blk = r >> 4;
  int chunk = blk & 1, ot = blk >> 1;
  int o = ot * 16 + m, c = chunk * 32 + kperm(s);
  dst[j] = (bf16)w[o * 64 + c];
}

// ---------------- per-step kernels ----------------

// concat[x_t, h] -> zero-padded NHWC bf16 (B,34,34,128)
__global__ __launch_bounds__(256) void k_pad_pack(const float* __restrict__ X,
                                                  const bf16* __restrict__ hpack,
                                                  bf16* __restrict__ padded, int t) {
  int i = blockIdx.x * 256 + threadIdx.x;     // < 16*34*34*128
  int cc = i & 127;
  int p = i >> 7;
  int xx = p % 34; p /= 34;
  int yy = p % 34;
  int b  = p / 34;
  bf16 v = (bf16)0.0f;
  if (yy > 0 && yy < 33 && xx > 0 && xx < 33) {
    int n = (yy - 1) * 32 + (xx - 1);
    if (cc < 64) v = (bf16)X[(((size_t)(b * 64 + cc)) * TT + t) * HWN + n];
    else         v = hpack[((size_t)b * HWN + n) * 64 + (cc - 64)];
  }
  padded[i] = v;
}

// implicit-GEMM 3x3 conv with 2x2 register blocking: one wave = 32 couts x 32 pixels.
// Per K-chunk: 2 A-frags + 2 B-frags (8 x b128) feed 4 WMMAs.
__global__ __launch_bounds__(256) void k_conv_gemm(const bf16* __restrict__ wpack,
                                                   const bf16* __restrict__ padded,
                                                   float* __restrict__ conv_out) {
  int wave = threadIdx.x >> 5, lane = threadIdx.x & 31;
  int l15 = lane & 15, hi = lane >> 4;
  int tile = blockIdx.x * 8 + wave;           // < 16*8*32 = 4096
  int b = tile >> 8, rem = tile & 255;
  int ctb = rem >> 5, ntb = rem & 31;         // cout-block(32), pixel-block(32)
  int n0 = ntb * 32 + l15, n1 = n0 + 16;
  int y0 = n0 >> 5, x0 = n0 & 31;
  int y1 = n1 >> 5, x1 = n1 & 31;

  // pull the 3 input rows each pixel column will walk toward the WGP caches
  const bf16* prow0 = padded + (((size_t)(b * 34 + y0)) * 34 + x0) * CCAT;
  const bf16* prow1 = padded + (((size_t)(b * 34 + y1)) * 34 + x1) * CCAT;
#pragma unroll
  for (int dy = 0; dy < 3; ++dy) {
    __builtin_prefetch(prow0 + (size_t)dy * 34 * CCAT, 0, 3);
    __builtin_prefetch(prow1 + (size_t)dy * 34 * CCAT, 0, 3);
  }

  f32x8 acc00 = {0.f,0.f,0.f,0.f,0.f,0.f,0.f,0.f};
  f32x8 acc01 = acc00, acc10 = acc00, acc11 = acc00;
  const bf16* wb0 = wpack + ((size_t)(ctb * 2) * NCHUNK) * 512 + l15 * 32 + hi * 16;
  const bf16* wb1 = wb0 + (size_t)NCHUNK * 512;

  for (int chunk = 0; chunk < NCHUNK; ++chunk) {
    bf16x16 a0 = ldfrag(wb0 + (size_t)chunk * 512);
    bf16x16 a1 = ldfrag(wb1 + (size_t)chunk * 512);
    int tap = chunk >> 2, dy = tap / 3, dx = tap - dy * 3;
    int cin0 = (chunk & 3) * 32 + hi * 16;
    bf16x16 b0 = ldfrag(padded + (((size_t)(b * 34 + y0 + dy)) * 34 + (x0 + dx)) * CCAT + cin0);
    bf16x16 b1 = ldfrag(padded + (((size_t)(b * 34 + y1 + dy)) * 34 + (x1 + dx)) * CCAT + cin0);
    acc00 = wmma_bf16(a0, b0, acc00);
    acc01 = wmma_bf16(a0, b1, acc01);
    acc10 = wmma_bf16(a1, b0, acc10);
    acc11 = wmma_bf16(a1, b1, acc11);
  }
  {
    float* op = conv_out + ((size_t)(b * G4 + (ctb * 2 + 0) * 16 + hi * 8)) * HWN + ntb * 32 + l15;
#pragma unroll
    for (int j = 0; j < 8; ++j) { op[(size_t)j * HWN] = acc00[j]; op[(size_t)j * HWN + 16] = acc01[j]; }
  }
  {
    float* op = conv_out + ((size_t)(b * G4 + (ctb * 2 + 1) * 16 + hi * 8)) * HWN + ntb * 32 + l15;
#pragma unroll
    for (int j = 0; j < 8; ++j) { op[(size_t)j * HWN] = acc10[j]; op[(size_t)j * HWN + 16] = acc11[j]; }
  }
}

// LSTM gates + peephole; updates c (f32), writes h_lstm as (B, n, 64) bf16 (GEMM B-layout)
__global__ __launch_bounds__(256) void k_gates(const float* __restrict__ conv_out,
                                               const float* __restrict__ conv_b,
                                               const float* __restrict__ Wci,
                                               const float* __restrict__ Wcf,
                                               const float* __restrict__ Wco,
                                               float* __restrict__ cbuf,
                                               bf16* __restrict__ hlstm) {
  int i = blockIdx.x * 256 + threadIdx.x;     // < 16*64*1024
  int n = i & 1023, ch = (i >> 10) & 63, b = i >> 16;
  size_t base = ((size_t)(b * G4 + ch) << 10) + n;
  const size_t gs = (size_t)64 << 10;
  float ic = conv_out[base]          + conv_b[ch];
  float fc = conv_out[base + gs]     + conv_b[ch + 64];
  float gc = conv_out[base + 2 * gs] + conv_b[ch + 128];
  float oc = conv_out[base + 3 * gs] + conv_b[ch + 192];
  int pidx = (ch << 10) + n;
  float c  = cbuf[i];
  float ig = sigm(ic + Wci[pidx] * c);
  float fg = sigm(fc + Wcf[pidx] * c);
  float nc = fg * c + ig * tanhf(gc);
  float og = sigm(oc + Wco[pidx] * nc);
  float nh = og * tanhf(nc);
  cbuf[i] = nc;
  hlstm[(((size_t)b << 10) + n) * 64 + ch] = (bf16)nh;
}

// generic 1x1-conv WMMA GEMM: O[o][n] = sum_c W[o][c]*In[n][c] + bias[o]
// One wave owns one 16-pixel column tile and ALL output-channel tiles (B-frag reuse).
// MODE 0: -> qpack (B,n,32) A-perm slots   MODE 1: -> kpack (B,n,32) identity
// MODE 2: -> vpack (B,64,1024)             MODE 3: -> d_out f32 slice t + hpack (B,n,64)
template <int OT, int MODE>
__global__ __launch_bounds__(256) void k_gemm1x1(const bf16* __restrict__ apack,
                                                 const bf16* __restrict__ inb,
                                                 const float* __restrict__ bias,
                                                 int t,
                                                 bf16* __restrict__ out_bf,
                                                 float* __restrict__ out_f32) {
  int wave = threadIdx.x >> 5, lane = threadIdx.x & 31;
  int l15 = lane & 15, hi = lane >> 4;
  int tile = blockIdx.x * 8 + wave;           // < 16*64 = 1024
  int b = tile >> 6, nt = tile & 63;
  f32x8 acc[OT];
#pragma unroll
  for (int ot = 0; ot < OT; ++ot) acc[ot] = (f32x8){0.f,0.f,0.f,0.f,0.f,0.f,0.f,0.f};
#pragma unroll
  for (int chunk = 0; chunk < 2; ++chunk) {
    bf16x16 bv = ldfrag(inb + ((size_t)(b * HWN + nt * 16 + l15)) * 64 + chunk * 32 + hi * 16);
#pragma unroll
    for (int ot = 0; ot < OT; ++ot) {
      bf16x16 a = ldfrag(apack + ((size_t)(ot * 2 + chunk)) * 512 + l15 * 32 + hi * 16);
      acc[ot] = wmma_bf16(a, bv, acc[ot]);
    }
  }
#pragma unroll
  for (int ot = 0; ot < OT; ++ot) {
#pragma unroll
    for (int j = 0; j < 8; ++j) {
      int o = ot * 16 + j + hi * 8;
      int n = nt * 16 + l15;
      float val = acc[ot][j] + bias[o];
      if (MODE == 0) {
        int s = o;                                          // kperm^-1 (identity)
        out_bf[((size_t)(b * HWN + n)) * 32 + s] = (bf16)val;
      } else if (MODE == 1) {
        out_bf[((size_t)(b * HWN + n)) * 32 + o] = (bf16)val;
      } else if (MODE == 2) {
        out_bf[((size_t)(b * 64 + o)) * HWN + n] = (bf16)val;
      } else {
        out_f32[((((size_t)(b * 64 + o)) * TT + t) << 10) + n] = val;
        out_bf[((size_t)(b * HWN + n)) * 64 + o] = (bf16)val;
      }
    }
  }
}

// flash-style attention: softmax(q^T k) @ v^T, never materializing (1024,1024)
__global__ __launch_bounds__(256) void k_attn(const bf16* __restrict__ qpack,
                                              const bf16* __restrict__ kpack,
                                              const bf16* __restrict__ vpack,
                                              bf16* __restrict__ zbuf) {
  __shared__ bf16 lds[8][512];                // per-wave 16x32 P tile
  int wave = threadIdx.x >> 5, lane = threadIdx.x & 31;
  int l15 = lane & 15, hi = lane >> 4;
  int tile = blockIdx.x * 8 + wave;           // < 16*64 = 1024
  int b = tile >> 6, nt = tile & 63;
  bf16* my = lds[wave];
  bf16x16 qf = ldfrag(qpack + ((size_t)(b * HWN + nt * 16 + l15)) * 32 + hi * 16);
  float M[8], L[8];
  f32x8 Z[4];
  f32x8 zero8 = {0.f,0.f,0.f,0.f,0.f,0.f,0.f,0.f};
#pragma unroll
  for (int j = 0; j < 8; ++j) { M[j] = -1e30f; L[j] = 0.f; }
#pragma unroll
  for (int c = 0; c < 4; ++c) Z[c] = zero8;

  for (int mc = 0; mc < 32; ++mc) {
    bf16x16 kf0 = ldfrag(kpack + ((size_t)(b * HWN + (mc * 2 + 0) * 16 + l15)) * 32 + hi * 16);
    f32x8 S0 = wmma_bf16(qf, kf0, zero8);
    bf16x16 kf1 = ldfrag(kpack + ((size_t)(b * HWN + (mc * 2 + 1) * 16 + l15)) * 32 + hi * 16);
    f32x8 S1 = wmma_bf16(qf, kf1, zero8);

#pragma unroll
    for (int j = 0; j < 8; ++j) {              // online softmax per row (rows live in lane-half)
      float mx = fmaxf(S0[j], S1[j]);
#pragma unroll
      for (int o = 8; o >= 1; o >>= 1) mx = fmaxf(mx, __shfl_xor(mx, o, 16));
      float nm = fmaxf(M[j], mx);
      float sc = __expf(M[j] - nm);
      float p0 = __expf(S0[j] - nm);
      float p1 = __expf(S1[j] - nm);
      float sm = p0 + p1;
#pragma unroll
      for (int o = 8; o >= 1; o >>= 1) sm += __shfl_xor(sm, o, 16);
      L[j] = L[j] * sc + sm;
      M[j] = nm;
      S0[j] = p0; S1[j] = p1;
      Z[0][j] *= sc; Z[1][j] *= sc; Z[2][j] *= sc; Z[3][j] *= sc;
    }
    // redistribute P: C/D layout -> A layout via LDS (store with kperm on columns)
    int row = hi * 8;
#pragma unroll
    for (int j = 0; j < 8; ++j) {
      my[(row + j) * 32 + kperm(l15)]      = (bf16)S0[j];
      my[(row + j) * 32 + kperm(16 + l15)] = (bf16)S1[j];
    }
    asm volatile("" ::: "memory");             // DS ops are in-order within a wave
    bf16x16 pf = ldfrag(my + l15 * 32 + hi * 16);
    asm volatile("" ::: "memory");
#pragma unroll
    for (int c = 0; c < 4; ++c) {
      bf16x16 vf = ldfrag(vpack + ((size_t)(b * 64 + c * 16 + l15)) * HWN + mc * 32 + hi * 16);
      Z[c] = wmma_bf16(pf, vf, Z[c]);
    }
  }
#pragma unroll
  for (int c = 0; c < 4; ++c)
#pragma unroll
    for (int j = 0; j < 8; ++j) {
      float z = Z[c][j] / L[j];
      zbuf[((size_t)(b * HWN + nt * 16 + j + hi * 8)) * 64 + c * 16 + l15] = (bf16)z;
    }
}

// ---------------- host launch ----------------

extern "C" void kernel_launch(void* const* d_in, const int* in_sizes, int n_in,
                              void* d_out, int out_size, void* d_ws, size_t ws_size,
                              hipStream_t stream) {
  (void)in_sizes; (void)n_in; (void)out_size; (void)ws_size;
  const float* X      = (const float*)d_in[0];
  const float* conv_w = (const float*)d_in[1];
  const float* conv_b = (const float*)d_in[2];
  const float* W_ci   = (const float*)d_in[3];
  const float* W_cf   = (const float*)d_in[4];
  const float* W_co   = (const float*)d_in[5];
  const float* q_w    = (const float*)d_in[6];
  const float* q_b    = (const float*)d_in[7];
  const float* k_w    = (const float*)d_in[8];
  const float* k_b    = (const float*)d_in[9];
  const float* v_w    = (const float*)d_in[10];
  const float* v_b    = (const float*)d_in[11];
  const float* z_w    = (const float*)d_in[12];
  const float* z_b    = (const float*)d_in[13];
  float* out = (float*)d_out;

  char* ws = (char*)d_ws;
  size_t off = 0;
  auto carve = [&](size_t bytes) -> char* {
    char* p = ws + off;
    off = (off + bytes + 255) & ~(size_t)255;
    return p;
  };
  bf16*  wpack  = (bf16*)carve((size_t)NCHUNK * 256 * 32 * 2);   // 576 KB
  bf16*  apq    = (bf16*)carve(2048);
  bf16*  apk    = (bf16*)carve(2048);
  bf16*  apv    = (bf16*)carve(8192);
  bf16*  apz    = (bf16*)carve(8192);
  bf16*  padded = (bf16*)carve((size_t)BB * 34 * 34 * CCAT * 2); // 4.7 MB
  float* convo  = (float*)carve((size_t)BB * G4 * HWN * 4);      // 16.8 MB
  float* cbuf   = (float*)carve((size_t)BB * 64 * HWN * 4);      // 4.2 MB
  bf16*  hpack  = (bf16*)carve((size_t)BB * HWN * 64 * 2);       // carried h (attn out)
  bf16*  hlstm  = (bf16*)carve((size_t)BB * HWN * 64 * 2);       // LSTM h before attn
  bf16*  qpack  = (bf16*)carve((size_t)BB * HWN * 32 * 2);
  bf16*  kpack  = (bf16*)carve((size_t)BB * HWN * 32 * 2);
  bf16*  vpack  = (bf16*)carve((size_t)BB * 64 * HWN * 2);
  bf16*  zbuf   = (bf16*)carve((size_t)BB * HWN * 64 * 2);

  // init state + zero-padded fragment slots (graph-capture safe)
  hipMemsetAsync(hpack, 0, (size_t)BB * HWN * 64 * 2, stream);
  hipMemsetAsync(cbuf,  0, (size_t)BB * 64 * HWN * 4, stream);
  hipMemsetAsync(qpack, 0, (size_t)BB * HWN * 32 * 2, stream);
  hipMemsetAsync(kpack, 0, (size_t)BB * HWN * 32 * 2, stream);

  k_prepack_convw<<<(NCHUNK * 256 * 32) / 256, 256, 0, stream>>>(conv_w, wpack);
  k_prepack_small<<<40, 256, 0, stream>>>(q_w, k_w, v_w, z_w, apq, apk, apv, apz);

  const int padBlocks  = (BB * 34 * 34 * CCAT) / 256;   // 9248
  const int convBlocks = (BB * 8 * 32) / 8;             // 512  (2x2-blocked waves)
  const int gateBlocks = (BB * 64 * HWN) / 256;         // 4096
  const int gemBlocks  = (BB * 64) / 8;                 // 128  (all 1x1 GEMMs)
  const int attBlocks  = (BB * 64) / 8;                 // 128

  for (int t = 0; t < TT; ++t) {
    k_pad_pack<<<padBlocks, 256, 0, stream>>>(X, hpack, padded, t);
    k_conv_gemm<<<convBlocks, 256, 0, stream>>>(wpack, padded, convo);
    k_gates<<<gateBlocks, 256, 0, stream>>>(convo, conv_b, W_ci, W_cf, W_co, cbuf, hlstm);
    k_gemm1x1<1, 0><<<gemBlocks, 256, 0, stream>>>(apq, hlstm, q_b, t, qpack, nullptr);
    k_gemm1x1<1, 1><<<gemBlocks, 256, 0, stream>>>(apk, hlstm, k_b, t, kpack, nullptr);
    k_gemm1x1<4, 2><<<gemBlocks, 256, 0, stream>>>(apv, hlstm, v_b, t, vpack, nullptr);
    k_attn<<<attBlocks, 256, 0, stream>>>(qpack, kpack, vpack, zbuf);
    k_gemm1x1<4, 3><<<gemBlocks, 256, 0, stream>>>(apz, zbuf, z_b, t, hpack, out);
  }
}
